// MMDLoss_39135742001578
// MI455X (gfx1250) — matrix operand order repaired and checked by reading, not compiled, not run on hardware
//
#include <hip/hip_runtime.h>

// ---------------------------------------------------------------------------
// MMD loss (RBF kernel) for gfx1250 / MI455X.
//   mean(exp(-g*d2(s,s))) + mean(exp(-g*d2(t,t))) - 2*mean(exp(-g*d2(s,t)))
// GEMM via v_wmma_f32_16x16x32_f16 with f16 hi/lo split (f32-accurate),
// fused exp + deterministic block reduction (no float atomics).
// Wave tile 64x64 (4x4 WMMA subtiles) -> 48 WMMAs per 32 b128 loads per
// K-step (~48 FLOP per L1 byte); inputs are L2-resident (16 MB << 192 MB L2).
// Symmetric terms (ss, tt) computed on the lower triangle only: blocks fully
// above the diagonal early-exit (write 0 partial); straddling/below blocks
// use per-element weights {0,1,2}. Exact and deterministic.
// ---------------------------------------------------------------------------

typedef _Float16 v16h __attribute__((ext_vector_type(16)));
typedef float    v8f  __attribute__((ext_vector_type(8)));
typedef float    v4f  __attribute__((ext_vector_type(4)));

#define NROWS 8192
#define DIMK  256
#define GAMMA_F 0.00390625f   // 1/256

union FragU { v16h h; v4f f[2]; };

// A-fragment (16x32 f16): lane holds 8 halves at kc and 8 halves at kc+16.
__device__ __forceinline__ v16h load_fragA(const _Float16* __restrict__ rowBase, int kc) {
  FragU u;
  u.f[0] = *reinterpret_cast<const v4f*>(rowBase + kc);
  u.f[1] = *reinterpret_cast<const v4f*>(rowBase + kc + 16);
  return u.h;
}
// B-fragment (32x16 f16, fed as rows of Y): lane holds 8 halves at kc and 8 halves at kc+16.
__device__ __forceinline__ v16h load_fragB(const _Float16* __restrict__ rowBase, int kc) {
  FragU u;
  u.f[0] = *reinterpret_cast<const v4f*>(rowBase + kc);
  u.f[1] = *reinterpret_cast<const v4f*>(rowBase + kc + 16);
  return u.h;
}

// ---------------------------------------------------------------------------
// Kernel 1: f32 -> (f16 hi, f16 lo) planes + exact f32 row norms.
// grid = (8192, 2), block = 256 (one block per row, blockIdx.y = matrix)
// ---------------------------------------------------------------------------
__global__ __launch_bounds__(256)
void mmd_prep_kernel(const float* __restrict__ zs, const float* __restrict__ zt,
                     _Float16* __restrict__ shi, _Float16* __restrict__ slo,
                     _Float16* __restrict__ thi, _Float16* __restrict__ tlo,
                     float* __restrict__ x2s, float* __restrict__ x2t) {
  __shared__ float smem[256];
  const int row = blockIdx.x;
  const int mat = blockIdx.y;
  const int tid = threadIdx.x;

  const float*   src = mat ? zt  : zs;
  _Float16*      hi  = mat ? thi : shi;
  _Float16*      lo  = mat ? tlo : slo;
  float*         x2  = mat ? x2t : x2s;

  const float v = src[(size_t)row * DIMK + tid];
  const _Float16 h = (_Float16)v;
  const float    r = v - (float)h;
  hi[(size_t)row * DIMK + tid] = h;
  lo[(size_t)row * DIMK + tid] = (_Float16)r;

  smem[tid] = v * v;
  __syncthreads();
  for (int s = 128; s > 0; s >>= 1) {
    if (tid < s) smem[tid] += smem[tid + s];
    __syncthreads();
  }
  if (tid == 0) x2[row] = smem[0];
}

// ---------------------------------------------------------------------------
// Kernel 2: tiled WMMA GEMM + fused exp + per-block reduction.
// grid = (64, 32, 3), block = 256 (8 waves). Block tile = 256(M) x 128(N).
// Wave tile = 64x64 (4 x 4 subtiles of 16x16). z selects pair: 0=ss 1=tt 2=st.
// ---------------------------------------------------------------------------
__global__ __launch_bounds__(256)
void mmd_tile_kernel(const _Float16* __restrict__ shi, const _Float16* __restrict__ slo,
                     const _Float16* __restrict__ thi, const _Float16* __restrict__ tlo,
                     const float* __restrict__ x2s, const float* __restrict__ x2t,
                     float* __restrict__ partials) {
  const int z = blockIdx.z;
  const int tid    = threadIdx.x;
  const int rowBlk = blockIdx.y * 256;
  const int colBlk = blockIdx.x * 128;
  const bool sym   = (z != 2);

  // Symmetric terms: blocks fully above the diagonal carry zero weight.
  // Whole block exits uniformly (EXEC stays full for the WMMA path otherwise).
  if (sym && colBlk >= rowBlk + 256) {
    if (tid == 0) partials[(size_t)z * 2048 + blockIdx.y * 64 + blockIdx.x] = 0.0f;
    return;
  }

  const _Float16* Ahi = (z == 1) ? thi : shi;
  const _Float16* Alo = (z == 1) ? tlo : slo;
  const _Float16* Bhi = (z == 0) ? shi : thi;
  const _Float16* Blo = (z == 0) ? slo : tlo;
  const float*    x2A = (z == 1) ? x2t : x2s;
  const float*    x2B = (z == 0) ? x2s : x2t;

  __shared__ float sX2[256];
  __shared__ float sY2[128];
  __shared__ float sRed[256];

  sX2[tid] = x2A[rowBlk + tid];
  if (tid < 128) sY2[tid] = x2B[colBlk + tid];
  __syncthreads();

  const int lane  = tid & 31;
  const int wave  = tid >> 5;
  const int waveM = wave & 3;   // 4 wave-rows * 64
  const int waveN = wave >> 2;  // 2 wave-cols * 64
  const int m0    = rowBlk + waveM * 64;
  const int n0    = colBlk + waveN * 64;
  const int lmod  = lane & 15;
  const int lhi   = lane >> 4;

  const _Float16* aHiBase = Ahi + (size_t)(m0 + lmod) * DIMK;
  const _Float16* aLoBase = Alo + (size_t)(m0 + lmod) * DIMK;
  const _Float16* bHiBase = Bhi + (size_t)(n0 + lmod) * DIMK;
  const _Float16* bLoBase = Blo + (size_t)(n0 + lmod) * DIMK;

  const int kaOff = lhi << 3;   // A: K-chunks {0..7,16..23} / {8..15,24..31}
  const int kbOff = lhi << 3;   // B: K-chunks {0..7,16..23} / {8..15,24..31}

  v8f acc[4][4] = {};

  for (int k0 = 0; k0 < DIMK; k0 += 32) {
    v16h ah[4], al[4];
#pragma unroll
    for (int mi = 0; mi < 4; mi++) {
      ah[mi] = load_fragA(aHiBase + (size_t)mi * 16 * DIMK, k0 + kaOff);
      al[mi] = load_fragA(aLoBase + (size_t)mi * 16 * DIMK, k0 + kaOff);
    }
#pragma unroll
    for (int ni = 0; ni < 4; ni++) {
      const v16h bh = load_fragB(bHiBase + (size_t)ni * 16 * DIMK, k0 + kbOff);
      const v16h bl = load_fragB(bLoBase + (size_t)ni * 16 * DIMK, k0 + kbOff);
#pragma unroll
      for (int mi = 0; mi < 4; mi++) {
        // hi*hi + hi*lo + lo*hi  (f32-accurate split product)
        acc[mi][ni] = __builtin_amdgcn_wmma_f32_16x16x32_f16(false, ah[mi], false, bh, (short)0, acc[mi][ni], false, false);
        acc[mi][ni] = __builtin_amdgcn_wmma_f32_16x16x32_f16(false, ah[mi], false, bl, (short)0, acc[mi][ni], false, false);
        acc[mi][ni] = __builtin_amdgcn_wmma_f32_16x16x32_f16(false, al[mi], false, bh, (short)0, acc[mi][ni], false, false);
      }
    }
  }

  // Fused epilogue: d2 = x2 + y2 - 2s, clamp, exp, weight, accumulate.
  // C/D layout: lane l -> N = l%16; VGPR v -> M = v + 8*(l/16).
  // Symmetric terms: weight 2 below diagonal, 1 on it, 0 above (exact).
  float part = 0.f;
#pragma unroll
  for (int mi = 0; mi < 4; mi++) {
    const int mloc = waveM * 64 + mi * 16 + (lhi << 3);
#pragma unroll
    for (int ni = 0; ni < 4; ni++) {
      const int nloc = waveN * 64 + ni * 16 + lmod;
      const float y2 = sY2[nloc];
      const int   ng = colBlk + nloc;
#pragma unroll
      for (int v = 0; v < 8; v++) {
        float d2 = sX2[mloc + v] + y2 - 2.0f * acc[mi][ni][v];
        d2 = fmaxf(d2, 0.0f);
        float e = __expf(-GAMMA_F * d2);
        if (sym) {
          const int mg = rowBlk + mloc + v;
          e = (ng > mg) ? 0.0f : ((ng < mg) ? 2.0f * e : e);
        }
        part += e;
      }
    }
  }

  sRed[tid] = part;
  __syncthreads();
  for (int s = 128; s > 0; s >>= 1) {
    if (tid < s) sRed[tid] += sRed[tid + s];
    __syncthreads();
  }
  if (tid == 0) partials[(size_t)z * 2048 + blockIdx.y * 64 + blockIdx.x] = sRed[0];
}

// ---------------------------------------------------------------------------
// Kernel 3: deterministic final reduction of 3 x 2048 partials.
// ---------------------------------------------------------------------------
__global__ __launch_bounds__(256)
void mmd_finalize_kernel(const float* __restrict__ partials, float* __restrict__ out) {
  __shared__ float s0[256], s1[256], s2[256];
  const int tid = threadIdx.x;
  float a = 0.f, b = 0.f, c = 0.f;
  for (int i = tid; i < 2048; i += 256) {
    a += partials[i];
    b += partials[2048 + i];
    c += partials[4096 + i];
  }
  s0[tid] = a; s1[tid] = b; s2[tid] = c;
  __syncthreads();
  for (int s = 128; s > 0; s >>= 1) {
    if (tid < s) { s0[tid] += s0[tid + s]; s1[tid] += s1[tid + s]; s2[tid] += s2[tid + s]; }
    __syncthreads();
  }
  if (tid == 0) {
    const float inv = 1.0f / (8192.0f * 8192.0f);
    out[0] = (s0[0] + s1[0] - 2.0f * s2[0]) * inv;
  }
}

// ---------------------------------------------------------------------------
extern "C" void kernel_launch(void* const* d_in, const int* in_sizes, int n_in,
                              void* d_out, int out_size, void* d_ws, size_t ws_size,
                              hipStream_t stream) {
  const float* zs = (const float*)d_in[0];
  const float* zt = (const float*)d_in[1];

  char* w = (char*)d_ws;
  const size_t planeBytes = (size_t)NROWS * DIMK * sizeof(_Float16); // 4 MB
  _Float16* shi = (_Float16*)(w + 0 * planeBytes);
  _Float16* slo = (_Float16*)(w + 1 * planeBytes);
  _Float16* thi = (_Float16*)(w + 2 * planeBytes);
  _Float16* tlo = (_Float16*)(w + 3 * planeBytes);
  float* x2s      = (float*)(w + 4 * planeBytes);
  float* x2t      = x2s + NROWS;
  float* partials = x2t + NROWS;   // 3 * 2048 floats

  mmd_prep_kernel<<<dim3(NROWS, 2), 256, 0, stream>>>(zs, zt, shi, slo, thi, tlo, x2s, x2t);
  mmd_tile_kernel<<<dim3(64, 32, 3), 256, 0, stream>>>(shi, slo, thi, tlo, x2s, x2t, partials);
  mmd_finalize_kernel<<<1, 256, 0, stream>>>(partials, (float*)d_out);
}
